// WaveFormer_65180423685205
// MI455X (gfx1250) — compile-verified, matrix operand order checked by reading, not run on hardware
//
#include <hip/hip_runtime.h>
#include <hip/hip_bf16.h>

typedef __attribute__((ext_vector_type(2)))  float  v2f;
typedef __attribute__((ext_vector_type(8)))  float  v8f;
typedef __attribute__((ext_vector_type(8)))  __bf16 v8bf;
typedef __attribute__((ext_vector_type(16))) __bf16 v16bf;

__device__ __forceinline__ unsigned lds_byte_off(const void* p) {
  // LDS aperture: low 32 bits of a flat shared-memory address are the LDS offset
  return (unsigned)(uintptr_t)p;
}

// ---------------------------------------------------------------------------
// Kernel 0: fp32 -> bf16 conversion (weights / freq_embed prep)
// ---------------------------------------------------------------------------
__global__ __launch_bounds__(256) void to_bf16_kernel(
    const float* __restrict__ src, __bf16* __restrict__ dst)
{
  long long i = (long long)blockIdx.x * 256 + threadIdx.x;
  dst[i] = (__bf16)src[i];
}

// ---------------------------------------------------------------------------
// Kernel 1: depthwise 3x3 conv (SAME, zero pad) + bias -> bf16 (b,c,hw)
// ---------------------------------------------------------------------------
__global__ __launch_bounds__(256) void dwconv_kernel(
    const float* __restrict__ x, const float* __restrict__ dw_w,
    const float* __restrict__ dw_b, __bf16* __restrict__ out)
{
  int g  = blockIdx.x * 256 + threadIdx.x;   // flat (b,c,h,w)
  int wq = g & 63, h = (g >> 6) & 63;
  int p  = g >> 12;                          // b*384 + c
  int c  = p % 384;
  const float* plane = x + (long long)p * 4096;
  const float* wk    = dw_w + c * 9;
  float acc = 0.f;
  #pragma unroll
  for (int ky = 0; ky < 3; ++ky) {
    int hh = h + ky - 1;
    if (hh < 0 || hh > 63) continue;
    #pragma unroll
    for (int kx = 0; kx < 3; ++kx) {
      int ww = wq + kx - 1;
      if (ww < 0 || ww > 63) continue;
      acc += plane[hh * 64 + ww] * wk[ky * 3 + kx];
    }
  }
  out[g] = (__bf16)(acc + dw_b[c]);
}

// ---------------------------------------------------------------------------
// Shared GEMM pieces.  LDS tiles are [row][k] with rows padded to 40 bf16
// (80 B, 16B-aligned) so fragment gathers are two 16B ds loads per operand.
// MMA: v_wmma_f32_16x16x32_bf16, 2x2 accumulators per wave (32x32/wave,
// 128x64 per 8-wave workgroup).
// ---------------------------------------------------------------------------
__device__ __forceinline__ void gemm_tile_mma(
    const __bf16* As, const __bf16* Bs, int lane, int wm, int wn, v8f acc[2][2])
{
  const int kh = (lane < 16) ? 0 : 8;   // k-half split per ISA 16-bit layout
  v16bf afrag[2], bfrag[2];
  #pragma unroll
  for (int t = 0; t < 2; ++t) {
    const __bf16* ra = &As[(wm + t * 16 + (lane & 15)) * 40];
    v8bf alo = *(const v8bf*)(ra + kh);
    v8bf ahi = *(const v8bf*)(ra + 16 + kh);
    afrag[t] = __builtin_shufflevector(alo, ahi, 0,1,2,3,4,5,6,7,
                                                 8,9,10,11,12,13,14,15);
    const __bf16* rb = &Bs[(wn + t * 16 + (lane & 15)) * 40];
    v8bf blo = *(const v8bf*)(rb + kh);
    v8bf bhi = *(const v8bf*)(rb + 16 + kh);
    bfrag[t] = __builtin_shufflevector(blo, bhi, 0,1,2,3,4,5,6,7,
                                                 8,9,10,11,12,13,14,15);
  }
  #pragma unroll
  for (int tm = 0; tm < 2; ++tm)
    #pragma unroll
    for (int tn = 0; tn < 2; ++tn)
      acc[tm][tn] = __builtin_amdgcn_wmma_f32_16x16x32_bf16(
          false, afrag[tm], false, bfrag[tn], (short)0, acc[tm][tn],
          false, false);
}

// issue one async 16B memory->LDS chunk (GVS mode, ASYNCcnt-tracked)
__device__ __forceinline__ void async_b128(unsigned loff, unsigned goff,
                                           const void* base)
{
  asm volatile("global_load_async_to_lds_b128 %0, %1, %2"
               :: "v"(loff), "v"(goff), "s"(base) : "memory");
}
__device__ __forceinline__ void wait_async0() {
  asm volatile("s_wait_asynccnt 0" ::: "memory");
}

// ---------------------------------------------------------------------------
// GEMM variant 1: both operands bf16, row-major, k-contiguous (row stride =
// K*2 bytes, 16B aligned).  All tile staging via global_load_async_to_lds.
// ---------------------------------------------------------------------------
template <typename Epi>
__global__ __launch_bounds__(256) void gemm_bf16_asyncA_kernel(
    const __bf16* __restrict__ A, const __bf16* __restrict__ Bm,
    int K, Epi epi)
{
  __shared__ __align__(16) __bf16 As[128 * 40];
  __shared__ __align__(16) __bf16 Bs[64 * 40];
  const int tid = threadIdx.x, lane = tid & 31, wave = tid >> 5;
  const int wm = (wave & 3) * 32, wn = (wave >> 2) * 32;
  const long long m0 = (long long)blockIdx.x * 128;
  const int n0 = blockIdx.y * 64;
  const unsigned aLds = lds_byte_off(As);
  const unsigned bLds = lds_byte_off(Bs);
  const unsigned Krow = (unsigned)K * 2u;     // row stride in bytes
  v8f acc[2][2] = {};

  for (int k0 = 0; k0 < K; k0 += 32) {
    #pragma unroll
    for (int it = 0; it < 2; ++it) {          // A: 128 rows x 4 x 16B
      int ci = tid + it * 256;
      int mm = ci >> 2, sub = ci & 3;
      unsigned loff = aLds + (unsigned)(mm * 80 + sub * 16);
      unsigned goff = (unsigned)(m0 + mm) * Krow + (unsigned)(k0 * 2 + sub * 16);
      async_b128(loff, goff, A);
    }
    {                                          // B: 64 rows x 4 x 16B
      int nn = tid >> 2, sub = tid & 3;
      unsigned loff = bLds + (unsigned)(nn * 80 + sub * 16);
      unsigned goff = (unsigned)(n0 + nn) * Krow + (unsigned)(k0 * 2 + sub * 16);
      async_b128(loff, goff, Bm);
    }
    wait_async0();
    __syncthreads();
    gemm_tile_mma(As, Bs, lane, wm, wn, acc);
    __syncthreads();
  }

  const int moff = (lane < 16) ? 0 : 8;
  const int nlan = lane & 15;
  #pragma unroll
  for (int tm = 0; tm < 2; ++tm)
    #pragma unroll
    for (int tn = 0; tn < 2; ++tn)
      #pragma unroll
      for (int r = 0; r < 8; ++r)
        epi(m0 + wm + tm * 16 + r + moff, n0 + wn + tn * 16 + nlan,
            acc[tm][tn][r]);
}

// ---------------------------------------------------------------------------
// GEMM variant 2: A is bf16 in (b,c,hw) layout (k-strided, m-contiguous):
//   addr = ((m>>12)*K + k)*4096 + (m&4095).  A staged synchronously with a
// transpose through VGPRs; B (weights) staged async.
// ---------------------------------------------------------------------------
template <typename Epi>
__global__ __launch_bounds__(256) void gemm_bf16_transA_kernel(
    const __bf16* __restrict__ A, const __bf16* __restrict__ Bm,
    int K, Epi epi)
{
  __shared__ __align__(16) __bf16 As[128 * 40];
  __shared__ __align__(16) __bf16 Bs[64 * 40];
  const int tid = threadIdx.x, lane = tid & 31, wave = tid >> 5;
  const int wm = (wave & 3) * 32, wn = (wave >> 2) * 32;
  const long long m0 = (long long)blockIdx.x * 128;
  const int n0 = blockIdx.y * 64;
  const unsigned bLds = lds_byte_off(Bs);
  const unsigned Krow = (unsigned)K * 2u;
  v8f acc[2][2] = {};

  for (int k0 = 0; k0 < K; k0 += 32) {
    {                                          // B async: 64 rows x 4 x 16B
      int nn = tid >> 2, sub = tid & 3;
      unsigned loff = bLds + (unsigned)(nn * 80 + sub * 16);
      unsigned goff = (unsigned)(n0 + nn) * Krow + (unsigned)(k0 * 2 + sub * 16);
      async_b128(loff, goff, Bm);
    }
    #pragma unroll
    for (int it = 0; it < 16; ++it) {          // A sync transpose: 32k x 128m
      int idx = tid + it * 256;
      int kk = idx >> 7, mm = idx & 127;       // consecutive tid -> contig m
      long long mg  = m0 + mm;
      long long off = ((mg >> 12) * (long long)K + (k0 + kk)) * 4096 +
                      (mg & 4095);
      As[mm * 40 + kk] = A[off];
    }
    wait_async0();
    __syncthreads();
    gemm_tile_mma(As, Bs, lane, wm, wn, acc);
    __syncthreads();
  }

  const int moff = (lane < 16) ? 0 : 8;
  const int nlan = lane & 15;
  #pragma unroll
  for (int tm = 0; tm < 2; ++tm)
    #pragma unroll
    for (int tn = 0; tn < 2; ++tn)
      #pragma unroll
      for (int r = 0; r < 8; ++r)
        epi(m0 + wm + tm * 16 + r + moff, n0 + wn + tn * 16 + nlan,
            acc[tm][tn][r]);
}

// ---- epilogues -------------------------------------------------------------
struct EpiLin {              // xz split: xs -> (b,c,hw) fp32 ; z -> (b,hw,c)
  float* xs; float* z; const float* lin_b;
  __device__ void operator()(long long m, int n, float acc) const {
    acc += lin_b[n];
    long long b = m >> 12, hw = m & 4095;
    if (n < 384) xs[(b * 384 + n) * 4096 + hw] = acc;
    else         z[m * 384 + (n - 384)]        = acc;
  }
};
struct EpiMult {             // mult[c][hw] = cos(c0 t) + sin(c0 t)(1+a/2)/c0
  float* mult; const float* tok_b; const float* cp; const float* ap;
  __device__ void operator()(long long m, int n, float acc) const {
    float c0 = cp[0];
    float t  = fmaxf(acc + tok_b[n], 0.f);
    float s1 = (1.f + 0.5f * ap[0]) / c0;
    mult[(long long)n * 4096 + m] = __cosf(c0 * t) + __sinf(c0 * t) * s1;
  }
};
struct EpiOut {              // out -> (b, C, hw) fp32
  float* outp; const float* out_b;
  __device__ void operator()(long long m, int n, float acc) const {
    long long b = m >> 12, hw = m & 4095;
    outp[(b * 384 + n) * 4096 + hw] = acc + out_b[n];
  }
};

// ---------------------------------------------------------------------------
// Kernel 4: per-(b,c) 64x64 DCT -> modulate -> inverse DCT, fp32-exact via
// V_WMMA_F32_16X16X4_F32.  One workgroup (8 waves) per image, all in LDS.
// ---------------------------------------------------------------------------
__device__ __forceinline__ v8f wmma_f32x4(v2f a, v2f b, v8f c) {
  return __builtin_amdgcn_wmma_f32_16x16x4_f32(false, a, false, b, (short)0,
                                               c, false, false);
}

template <bool AT, bool BT>
__device__ __forceinline__ void mm64(const float* __restrict__ A,
                                     const float* __restrict__ B,
                                     float* __restrict__ D, int lane, int wave)
{
  #pragma unroll
  for (int i = 0; i < 2; ++i) {                  // 16 tiles / 8 waves
    int t  = wave * 2 + i;
    int tr = (t >> 2) * 16, tc = (t & 3) * 16;
    v8f acc = {};
    const int kh = (lane < 16) ? 0 : 2;          // f32 A/B half-wave k split
    const int ml = tr + (lane & 15);
    const int nl = tc + (lane & 15);
    #pragma unroll
    for (int s = 0; s < 16; ++s) {               // K=64 in steps of 4
      int kb = s * 4 + kh;
      v2f a, b;
      if (!AT) a = *(const v2f*)&A[ml * 66 + kb];
      else { a.x = A[kb * 66 + ml]; a.y = A[(kb + 1) * 66 + ml]; }
      if (!BT) { b.x = B[kb * 66 + nl]; b.y = B[(kb + 1) * 66 + nl]; }
      else b = *(const v2f*)&B[nl * 66 + kb];
      acc = wmma_f32x4(a, b, acc);
    }
    const int moff = (lane < 16) ? 0 : 8;
    #pragma unroll
    for (int r = 0; r < 8; ++r)
      D[(tr + r + moff) * 66 + nl] = acc[r];
  }
}

__global__ __launch_bounds__(256) void dct_kernel(
    const float* __restrict__ xs, const float* __restrict__ mult,
    float* __restrict__ y)
{
  __shared__ __align__(16) float Wd[64 * 66];
  __shared__ __align__(16) float Xs[64 * 66];
  __shared__ __align__(16) float Ts[64 * 66];
  const int tid = threadIdx.x, lane = tid & 31, wave = tid >> 5;
  const int p = blockIdx.x;                      // b*384 + c
  const int c = p % 384;
  const float* src = xs + (long long)p * 4096;

  for (int idx = tid; idx < 4096; idx += 256) {  // DCT-II basis + load X
    int n = idx >> 6, xq = idx & 63;
    float v = __cosf((float)n * ((float)xq + 0.5f) * 0.04908738521f) *
              0.1767766953f;                     // pi/64, sqrt(2/64)
    if (n == 0) v *= 0.7071067812f;
    Wd[n * 66 + xq] = v;
    Xs[n * 66 + xq] = src[idx];
  }
  __syncthreads();
  mm64<false, false>(Wd, Xs, Ts, lane, wave); __syncthreads();  // T = W X
  mm64<false, true >(Ts, Wd, Xs, lane, wave); __syncthreads();  // Xd = T W^T
  const float* mp = mult + (long long)c * 4096;
  for (int idx = tid; idx < 4096; idx += 256)                    // modulate
    Xs[(idx >> 6) * 66 + (idx & 63)] *= mp[idx];
  __syncthreads();
  mm64<true , false>(Wd, Xs, Ts, lane, wave); __syncthreads();  // T2 = W^T Xd'
  mm64<false, false>(Ts, Wd, Xs, lane, wave); __syncthreads();  // xi = T2 W

  const long long b = p / 384;
  for (int idx = tid; idx < 4096; idx += 256)                    // y:(b,hw,c)
    y[(b * 4096 + idx) * 384 + c] = Xs[(idx >> 6) * 66 + (idx & 63)];
}

// ---------------------------------------------------------------------------
// Kernel 5: LayerNorm over C=384 + silu(z) gate -> bf16 (b,hw,c).
// One wave32 per pixel, 12 channels/lane, shfl_xor reductions.
// ---------------------------------------------------------------------------
__global__ __launch_bounds__(256) void ln_silu_kernel(
    const float* __restrict__ y, const float* __restrict__ z,
    const float* __restrict__ ln_g, const float* __restrict__ ln_b,
    __bf16* __restrict__ y2)
{
  const int lane = threadIdx.x & 31;
  const long long pix = (long long)blockIdx.x * 8 + (threadIdx.x >> 5);
  const float* row = y + pix * 384;
  float v[12], s = 0.f, s2 = 0.f;
  #pragma unroll
  for (int i = 0; i < 12; ++i) {
    v[i] = row[i * 32 + lane]; s += v[i]; s2 += v[i] * v[i];
  }
  #pragma unroll
  for (int off = 16; off > 0; off >>= 1) {
    s  += __shfl_xor(s,  off, 32);
    s2 += __shfl_xor(s2, off, 32);
  }
  float mu  = s * (1.f / 384.f);
  float var = s2 * (1.f / 384.f) - mu * mu;
  float r   = rsqrtf(var + 1e-5f);
  const float* zr = z + pix * 384;
  __bf16* o = y2 + pix * 384;
  #pragma unroll
  for (int i = 0; i < 12; ++i) {
    int ch = i * 32 + lane;
    float val = (v[i] - mu) * r * ln_g[ch] + ln_b[ch];
    float zz  = zr[ch];
    o[ch] = (__bf16)(val * (zz / (1.f + __expf(-zz))));
  }
}

// ---------------------------------------------------------------------------
extern "C" void kernel_launch(void* const* d_in, const int* in_sizes, int n_in,
                              void* d_out, int out_size, void* d_ws, size_t ws_size,
                              hipStream_t stream)
{
  (void)in_sizes; (void)n_in; (void)out_size; (void)ws_size;
  const float* x     = (const float*)d_in[0];
  const float* fe    = (const float*)d_in[1];
  const float* dw_w  = (const float*)d_in[2];
  const float* dw_b  = (const float*)d_in[3];
  const float* lin_w = (const float*)d_in[4];
  const float* lin_b = (const float*)d_in[5];
  const float* tok_w = (const float*)d_in[6];
  const float* tok_b = (const float*)d_in[7];
  const float* cp    = (const float*)d_in[8];
  const float* ap    = (const float*)d_in[9];
  const float* ln_g  = (const float*)d_in[10];
  const float* ln_b  = (const float*)d_in[11];
  const float* out_w = (const float*)d_in[12];
  const float* out_b = (const float*)d_in[13];
  float* outp = (float*)d_out;

  char* W = (char*)d_ws;
  __bf16* xdw  = (__bf16*)(W);                  // 25,165,824 B  (b,c,hw) bf16
  float*  xs   = (float*)(W + 25165824);        // 50,331,648 B  (b,c,hw) fp32
  float*  z    = (float*)(W + 75497472);        // 50,331,648 B  (b,hw,c) fp32
  float*  y    = (float*)(W + 125829120);       // 50,331,648 B  (b,hw,c) fp32
  float*  mult = (float*)(W + 176160768);       //  6,291,456 B  (c,hw)   fp32
  __bf16* feb  = (__bf16*)(W + 182452224);      //  3,145,728 B
  __bf16* linb = (__bf16*)(W + 185597952);      //    589,824 B
  __bf16* tokb = (__bf16*)(W + 186187776);      //    294,912 B
  __bf16* outw = (__bf16*)(W + 186482688);      //    294,912 B
  __bf16* y2   = (__bf16*)(W + 25165824);       // aliases xs (dead after DCT)

  // bf16 prep (weights + freq_embed)
  to_bf16_kernel<<<6144, 256, 0, stream>>>(fe,    feb);
  to_bf16_kernel<<<1152, 256, 0, stream>>>(lin_w, linb);
  to_bf16_kernel<<< 576, 256, 0, stream>>>(tok_w, tokb);
  to_bf16_kernel<<< 576, 256, 0, stream>>>(out_w, outw);

  dwconv_kernel<<<49152, 256, 0, stream>>>(x, dw_w, dw_b, xdw);

  EpiLin e1{xs, z, lin_b};                      // xz = x_dw @ lin_w^T
  gemm_bf16_transA_kernel<EpiLin><<<dim3(256, 12), 256, 0, stream>>>(
      xdw, linb, 384, e1);

  EpiMult e2{mult, tok_b, cp, ap};              // t = relu(fe @ tok_w^T)
  gemm_bf16_asyncA_kernel<EpiMult><<<dim3(32, 6), 256, 0, stream>>>(
      feb, tokb, 384, e2);

  dct_kernel<<<3072, 256, 0, stream>>>(xs, mult, y);

  ln_silu_kernel<<<4096, 256, 0, stream>>>(y, z, ln_g, ln_b, y2);

  EpiOut e3{outp, out_b};                       // out = y2 @ out_w^T
  gemm_bf16_asyncA_kernel<EpiOut><<<dim3(256, 6), 256, 0, stream>>>(
      y2, outw, 384, e3);
}
